// Model_17085379903581
// MI455X (gfx1250) — matrix operand order repaired and checked by reading, not compiled, not run on hardware
//
#include <hip/hip_runtime.h>

// ---------------------------------------------------------------------------
// S4D forecaster forward pass for MI455X (gfx1250, wave32, WMMA)
// B=32, L=1024, E=H=512, P=336, NLAYERS=3, N=32 (complex diag states)
// GEMM LDS tiles are stored in WMMA *fragment order* so all fragment loads are
// ds_load_b128 (no u16 gathers / v_mov_b16 merge chains).
// ---------------------------------------------------------------------------

#define Bsz 32
#define Lsz 1024
#define Hsz 512
#define Psz 336
#define Nst 32
#define NLAY 3

typedef __bf16 v16bf __attribute__((ext_vector_type(16)));
typedef float  v8f   __attribute__((ext_vector_type(8)));

union Frag {
  v16bf v;
  __bf16 h[16];
  unsigned u[8];
  uint4 q[2];
};

union H8 { uint4 v; __bf16 h[8]; };

__device__ __forceinline__ __bf16 f2bf(float x) {
  union { float f; unsigned u; } a; a.f = x;
  unsigned r = a.u + 0x7FFFu + ((a.u >> 16) & 1u);   // round-to-nearest-even
  union { unsigned short s; __bf16 b; } o; o.s = (unsigned short)(r >> 16);
  return o.b;
}

__device__ __forceinline__ float sigmoidf(float x) {
  return 1.0f / (1.0f + __expf(-x));
}

__device__ __forceinline__ float gelu_tanh(float v) {
  float v3 = v * v * v;
  return 0.5f * v * (1.0f + tanhf(0.7978845608028654f * (v + 0.044715f * v3)));
}

// ---------------------------------------------------------------------------
// (B,L,E) -> (B,H,L) transpose into f32 workspace
// ---------------------------------------------------------------------------
__global__ void transpose_in(const float* __restrict__ xe, float* __restrict__ u) {
  int idx = blockIdx.x * 256 + threadIdx.x;          // (b,h,l), l fastest
  int l = idx & (Lsz - 1);
  int h = (idx >> 10) & (Hsz - 1);
  int b = idx >> 19;
  u[idx] = xe[((size_t)(b * Lsz + l)) * Hsz + h];
}

// ---------------------------------------------------------------------------
// f32 -> bf16 elementwise convert
// ---------------------------------------------------------------------------
__global__ void cvt_bf(const float* __restrict__ src, __bf16* __restrict__ dst, int n) {
  int i = blockIdx.x * 256 + threadIdx.x;
  if (i < n) dst[i] = f2bf(src[i]);
}

// ---------------------------------------------------------------------------
// Per-layer SSM table generation. Thread = (h, n).
//   w = exp(dt*A);  Ct = C * expm1(dt*A)/A
//   W32[h,n]           = w^32
//   Pw[(h*32+j)*32+n]  = w^(31-j)                 (state input matrix)
//   Mcar[(h*64+n)*32+t]    =  2*Re(Ct*w^(t+1))    (carry -> output, Sre part)
//   Mcar[(h*64+32+n)*32+t] = -2*Im(Ct*w^(t+1))    (carry -> output, Sim part)
// ---------------------------------------------------------------------------
__global__ void s4_tables(const float* __restrict__ log_dt, const float* __restrict__ A_re,
                          const float* __restrict__ A_im,  const float* __restrict__ C_re,
                          const float* __restrict__ C_im,
                          float* __restrict__ Mcar, float2* __restrict__ Pw,
                          float2* __restrict__ W32) {
  int idx = blockIdx.x * 256 + threadIdx.x;          // H*N threads
  int h = idx >> 5, n = idx & 31;
  float dt  = __expf(log_dt[h]);
  float are = A_re[idx], aim = A_im[idx];
  float dre = dt * are, dimg = dt * aim;
  float er = __expf(dre), co = __cosf(dimg), si = __sinf(dimg);
  float em_re = er * co - 1.0f, em_im = er * si;     // expm1(dt*A)
  float den = are * are + aim * aim;
  float bre = (em_re * are + em_im * aim) / den;     // expm1(dtA)/A
  float bim = (em_im * are - em_re * aim) / den;
  float cre = C_re[idx], cim = C_im[idx];
  float ctr = cre * bre - cim * bim;                 // Ct
  float cti = cre * bim + cim * bre;

  float e32 = __expf(32.0f * dre);
  W32[idx] = make_float2(e32 * __cosf(32.0f * dimg), e32 * __sinf(32.0f * dimg));

  #pragma unroll 4
  for (int j = 0; j < 32; ++j) {
    float p = (float)(31 - j);
    float ep = __expf(p * dre);
    Pw[(h * 32 + j) * 32 + n] = make_float2(ep * __cosf(p * dimg), ep * __sinf(p * dimg));
  }
  #pragma unroll 4
  for (int t = 0; t < 32; ++t) {
    float p = (float)(t + 1);
    float ep = __expf(p * dre);
    float wr = ep * __cosf(p * dimg), wi = ep * __sinf(p * dimg);
    float zr = ctr * wr - cti * wi;
    float zi = ctr * wi + cti * wr;
    Mcar[(h * 64 + n) * 32 + t]      =  2.0f * zr;
    Mcar[(h * 64 + 32 + n) * 32 + t] = -2.0f * zi;
  }
}

// ---------------------------------------------------------------------------
// K table: K[h,t] = 2*sum_n Re(Ct_n * w_n^t), t in [0,32). Thread = (h,t).
// ---------------------------------------------------------------------------
__global__ void s4_ktab(const float* __restrict__ log_dt, const float* __restrict__ A_re,
                        const float* __restrict__ A_im,  const float* __restrict__ C_re,
                        const float* __restrict__ C_im,  float* __restrict__ Ktab) {
  int idx = blockIdx.x * 256 + threadIdx.x;          // H*32 threads
  int h = idx >> 5, t = idx & 31;
  float dt = __expf(log_dt[h]);
  float acc = 0.0f;
  for (int n = 0; n < 32; ++n) {
    int e = h * 32 + n;
    float are = A_re[e], aim = A_im[e];
    float dre = dt * are, dimg = dt * aim;
    float er = __expf(dre), co = __cosf(dimg), si = __sinf(dimg);
    float em_re = er * co - 1.0f, em_im = er * si;
    float den = are * are + aim * aim;
    float bre = (em_re * are + em_im * aim) / den;
    float bim = (em_im * are - em_re * aim) / den;
    float cre = C_re[e], cim = C_im[e];
    float ctr = cre * bre - cim * bim;
    float cti = cre * bim + cim * bre;
    float p = (float)t;
    float ep = __expf(p * dre);
    float wr = ep * __cosf(p * dimg), wi = ep * __sinf(p * dimg);
    acc += ctr * wr - cti * wi;
  }
  Ktab[h * 32 + t] = 2.0f * acc;
}

// ---------------------------------------------------------------------------
// Chunked SSM scan + D-skip + GELU -> bf16. One wave32 per (b,h) sequence.
// Lane t = time-in-chunk for output; lane n = complex state n for carry.
// 32 chunks of 32 steps. Writes gelu(y) as bf16 (GEMM input).
// ---------------------------------------------------------------------------
__global__ void s4_scan(const float* __restrict__ U, const float* __restrict__ Ktab,
                        const float* __restrict__ Mcar, const float2* __restrict__ Pw,
                        const float2* __restrict__ W32, const float* __restrict__ Dskip,
                        __bf16* __restrict__ Ybf) {
  __shared__ float sm[8][132];   // per wave: [0,32)=u  [32,64)=Sre  [64,96)=Sim  [96,128)=K
  int tid = threadIdx.x, lane = tid & 31, wid = tid >> 5;
  int seq = blockIdx.x * 8 + wid;
  int b = seq >> 9, h = seq & (Hsz - 1);

  float* uch = &sm[wid][0];
  float* Sre = &sm[wid][32];
  float* Sim = &sm[wid][64];
  float* Kl  = &sm[wid][96];
  Kl[lane] = Ktab[h * 32 + lane];

  float2 w32 = W32[h * 32 + lane];
  float dsk = Dskip[h];
  float sre = 0.0f, sim = 0.0f;

  const float*  base_u = U   + (size_t)(b * Hsz + h) * Lsz;
  __bf16*       base_y = Ybf + (size_t)(b * Hsz + h) * Lsz;
  const float*  Mrow   = Mcar + (size_t)h * 2048 + lane;   // Mcar[(h*64+k)*32 + t]
  const float2* Prow   = Pw   + (size_t)h * 1024 + lane;   // Pw[(h*32+j)*32 + n]

  for (int c = 0; c < 32; ++c) {
    float u = base_u[c * 32 + lane];
    uch[lane] = u;
    Sre[lane] = sre;
    Sim[lane] = sim;
    __syncthreads();

    // local causal Toeplitz: y[t] = sum_{j<=t} K[t-j]*u[j]
    float y = 0.0f;
    #pragma unroll 8
    for (int j = 0; j < 32; ++j) {
      int d = lane - j;
      float kv = Kl[d & 31];
      y += (d >= 0 ? kv : 0.0f) * uch[j];
    }
    // carry-in from state: y[t] += sum_k Mcar[h][k][t] * S[k]
    #pragma unroll 8
    for (int k = 0; k < 32; ++k) {
      y += Mrow[k * 32]        * Sre[k];
      y += Mrow[(32 + k) * 32] * Sim[k];
    }
    // state input accumulation (lane = state n): sum_j w^(31-j) * u[j]
    float ar = 0.0f, ai = 0.0f;
    #pragma unroll 8
    for (int j = 0; j < 32; ++j) {
      float2 p = Prow[j * 32];
      float uj = uch[j];
      ar += p.x * uj;
      ai += p.y * uj;
    }
    __syncthreads();

    float nsre = w32.x * sre - w32.y * sim + ar;
    float nsim = w32.x * sim + w32.y * sre + ai;
    sre = nsre; sim = nsim;

    float v = y + dsk * u;
    base_y[c * 32 + lane] = f2bf(gelu_tanh(v));
  }
}

// ---------------------------------------------------------------------------
// Fragment-order LDS helpers.
// A (16x32 bf16, M x K) fragment of lane: element e (dword) holds halves
//   k = 2e + 8*hi + 8*(e>=4), m = lane&15, hi = lane>>4.
// An 8-half col-block c (k in [8c,8c+8)) of row m == contiguous dwords
//   q = c>>1 (e-quad), lane = (m&15) + 16*(c&1). So global b128 -> LDS b128.
// B (32x16 bf16, K x N) fragment of lane: element half i holds
//   k = (i&7) + 8*hi + 16*(i>>3), n = lane&15 (same k order as A).
// ---------------------------------------------------------------------------

// ---------------------------------------------------------------------------
// GLU projection via WMMA bf16: z = Wo @ y + bo ; out = z_top * sigmoid(z_bot)
// Grid: (L/64, H/64, B). 256 threads = 8 waves; each wave: 1 M-subtile (top+bot
// accumulators) x 2 N-subtiles. K loop over H in 32-steps, LDS-staged in
// fragment order: all fragment reads are ds_load_b128.
// ---------------------------------------------------------------------------
__global__ void glu_gemm(const __bf16* __restrict__ Wo_bf,  // (2H, H)
                         const __bf16* __restrict__ Y,      // (B,H,L) bf16
                         const float* __restrict__ bo,      // (2H)
                         float* __restrict__ Xn) {          // (B,H,L) f32
  __shared__ uint4 sAt4[4 * 32 * 2];                 // 4 M-subtiles, frag order
  __shared__ uint4 sAb4[4 * 32 * 2];
  __shared__ __align__(16) __bf16 sBf[4 * 32 * 16];  // 4 N-subtiles, frag order

  int b = blockIdx.z, h0 = blockIdx.y * 64, l0 = blockIdx.x * 64;
  int tid = threadIdx.x, lane = tid & 31, wid = tid >> 5;
  int mw = wid & 3;
  int nw0 = (wid >> 2) * 2;
  int mm = lane & 15, hi = lane >> 4;

  // staging decompositions (loop-invariant)
  int idxA = tid * 8;
  int rowA = idxA >> 5, colblk = (idxA & 31) >> 3;   // 64 rows x 4 col-blocks
  int qA = colblk >> 1;
  int laneA = (rowA & 15) + 16 * (colblk & 1);
  int mwA = rowA >> 4;
  int slotA = (mwA * 32 + laneA) * 2 + qA;
  const __bf16* gAt = Wo_bf + (size_t)(h0 + rowA) * Hsz + colblk * 8;
  const __bf16* gAb = Wo_bf + (size_t)(Hsz + h0 + rowA) * Hsz + colblk * 8;

  int idxB = tid * 8;
  int rowK = idxB >> 6, colN = idxB & 63;            // 32 k-rows x 64 n-cols
  int iB = (rowK & 7) + 8 * (rowK >> 4), hiB = (rowK >> 3) & 1;
  int baseB = ((colN >> 4) * 32 + (colN & 15) + 16 * hiB) * 16 + iB;
  const __bf16* gB = Y + (size_t)(b * Hsz + rowK) * Lsz + l0 + colN;

  v8f acc[2][2];
  #pragma unroll
  for (int a = 0; a < 2; ++a)
    #pragma unroll
    for (int c = 0; c < 2; ++c)
      #pragma unroll
      for (int r = 0; r < 8; ++r) acc[a][c][r] = 0.0f;

  for (int kb = 0; kb < Hsz / 32; ++kb) {
    int k0 = kb * 32;
    // stage A top/bot: b128 global -> b128 LDS (fragment order)
    sAt4[slotA] = *(const uint4*)(gAt + k0);
    sAb4[slotA] = *(const uint4*)(gAb + k0);
    // stage B: b128 global read, 8 scattered b16 LDS stores (y contiguous in n)
    {
      H8 v; v.v = *(const uint4*)(gB + (size_t)k0 * Lsz);
      #pragma unroll
      for (int j = 0; j < 8; ++j) sBf[baseB + j * 16] = v.h[j];
    }
    __syncthreads();

    Frag at, ab;
    at.q[0] = sAt4[(mw * 32 + lane) * 2 + 0];
    at.q[1] = sAt4[(mw * 32 + lane) * 2 + 1];
    ab.q[0] = sAb4[(mw * 32 + lane) * 2 + 0];
    ab.q[1] = sAb4[(mw * 32 + lane) * 2 + 1];
    #pragma unroll
    for (int nb = 0; nb < 2; ++nb) {
      Frag bf;
      int lb = ((nw0 + nb) * 32 + lane) * 16;
      bf.q[0] = *(const uint4*)&sBf[lb];
      bf.q[1] = *(const uint4*)&sBf[lb + 8];
      acc[0][nb] = __builtin_amdgcn_wmma_f32_16x16x32_bf16(
          false, at.v, false, bf.v, (short)0, acc[0][nb], false, false);
      acc[1][nb] = __builtin_amdgcn_wmma_f32_16x16x32_bf16(
          false, ab.v, false, bf.v, (short)0, acc[1][nb], false, false);
    }
    __syncthreads();
  }

  // epilogue: bias + GLU gate, write f32 next-layer input
  #pragma unroll
  for (int nb = 0; nb < 2; ++nb) {
    int n = l0 + (nw0 + nb) * 16 + mm;
    #pragma unroll
    for (int r = 0; r < 8; ++r) {
      int mg = h0 + mw * 16 + r + 8 * hi;            // global channel (row)
      float zt = acc[0][nb][r] + bo[mg];
      float zb = acc[1][nb][r] + bo[Hsz + mg];
      Xn[((size_t)(b * Hsz) + mg) * Lsz + n] = zt * sigmoidf(zb);
    }
  }
}

// ---------------------------------------------------------------------------
// Output projection via WMMA bf16: out[b,p,h] = sum_l Wout[p,l]*x[b,h,l]+b_out[p]
// A = Wout (M=p), B = x^T (N=h) so stores over h are lane-coalesced.
// x is contiguous along k(=l), so B staging is b128 -> b128 (fragment order).
// Grid: (H/64, ceil(P/64), B).
// ---------------------------------------------------------------------------
__global__ void out_gemm(const __bf16* __restrict__ Wout_bf, // (P,L)
                         const __bf16* __restrict__ Xbf,     // (B,H,L)
                         const float* __restrict__ b_out,    // (P)
                         float* __restrict__ Out) {          // (B,P,E)
  __shared__ uint4 sA4[4 * 32 * 2];
  __shared__ uint4 sB4[4 * 32 * 2];

  int b = blockIdx.z, p0 = blockIdx.y * 64, h0 = blockIdx.x * 64;
  int tid = threadIdx.x, lane = tid & 31, wid = tid >> 5;
  int mw = wid & 3;
  int nw0 = (wid >> 2) * 2;
  int mm = lane & 15, hi = lane >> 4;

  // A staging decomposition
  int idxA = tid * 8;
  int rowA = idxA >> 5, colblk = (idxA & 31) >> 3;
  int slotA = ((rowA >> 4) * 32 + (rowA & 15) + 16 * (colblk & 1)) * 2 + (colblk >> 1);
  bool okA = (p0 + rowA) < Psz;
  const __bf16* gA = Wout_bf + (size_t)(p0 + rowA) * Lsz + colblk * 8;

  // B staging decomposition: sB[k][n] = x[b, h0+n, k0+k]; contiguous along k
  int nB = tid >> 2, kc = (tid & 3) * 8;             // 64 n x 4 k-blocks of 8
  int slotB = ((nB >> 4) * 32 + (nB & 15) + 16 * ((kc >> 3) & 1)) * 2 + (kc >> 4);
  const __bf16* gB = Xbf + (size_t)(b * Hsz + h0 + nB) * Lsz + kc;

  v8f acc[2];
  #pragma unroll
  for (int c = 0; c < 2; ++c)
    #pragma unroll
    for (int r = 0; r < 8; ++r) acc[c][r] = 0.0f;

  for (int kb = 0; kb < Lsz / 32; ++kb) {
    int k0 = kb * 32;
    uint4 va = make_uint4(0u, 0u, 0u, 0u);
    if (okA) va = *(const uint4*)(gA + k0);
    sA4[slotA] = va;
    sB4[slotB] = *(const uint4*)(gB + k0);
    __syncthreads();

    Frag a;
    a.q[0] = sA4[(mw * 32 + lane) * 2 + 0];
    a.q[1] = sA4[(mw * 32 + lane) * 2 + 1];
    #pragma unroll
    for (int nb = 0; nb < 2; ++nb) {
      Frag bf;
      bf.q[0] = sB4[((nw0 + nb) * 32 + lane) * 2 + 0];
      bf.q[1] = sB4[((nw0 + nb) * 32 + lane) * 2 + 1];
      acc[nb] = __builtin_amdgcn_wmma_f32_16x16x32_bf16(
          false, a.v, false, bf.v, (short)0, acc[nb], false, false);
    }
    __syncthreads();
  }

  #pragma unroll
  for (int nb = 0; nb < 2; ++nb) {
    int hcol = h0 + (nw0 + nb) * 16 + mm;
    #pragma unroll
    for (int r = 0; r < 8; ++r) {
      int p = p0 + mw * 16 + r + 8 * hi;
      if (p < Psz)
        Out[((size_t)b * Psz + p) * Hsz + hcol] = acc[nb][r] + b_out[p];
    }
  }
}

// ---------------------------------------------------------------------------
// Host-side orchestration (graph-capture safe: kernels only, all on stream)
// ---------------------------------------------------------------------------
extern "C" void kernel_launch(void* const* d_in, const int* in_sizes, int n_in,
                              void* d_out, int out_size, void* d_ws, size_t ws_size,
                              hipStream_t stream) {
  (void)in_sizes; (void)n_in; (void)out_size; (void)ws_size;

  const float* x_enc  = (const float*)d_in[0];
  const float* log_dt = (const float*)d_in[4];
  const float* A_re   = (const float*)d_in[5];
  const float* A_im   = (const float*)d_in[6];
  const float* C_re   = (const float*)d_in[7];
  const float* C_im   = (const float*)d_in[8];
  const float* Dskip  = (const float*)d_in[9];
  const float* Wo     = (const float*)d_in[10];
  const float* bo     = (const float*)d_in[11];
  const float* W_out  = (const float*)d_in[12];
  const float* b_out  = (const float*)d_in[13];
  float* Out = (float*)d_out;

  // workspace layout (bytes, 256B-aligned offsets); total ~172 MB
  char* ws = (char*)d_ws;
  float*  bufA    = (float*) (ws + 0);           // B*H*L f32   = 64 MB
  float*  bufB    = (float*) (ws + 67108864);    // B*H*L f32   = 64 MB
  __bf16* ybf     = (__bf16*)(ws + 134217728);   // B*H*L bf16  = 32 MB
  __bf16* wo_bf   = (__bf16*)(ws + 167772160);   // 3*2H*H bf16 = 3 MB
  __bf16* wout_bf = (__bf16*)(ws + 170917888);   // P*L bf16    = 0.66 MB
  float*  ktab    = (float*) (ws + 171606016);   // H*32 f32
  float*  mcar    = (float*) (ws + 171671552);   // H*64*32 f32 = 4 MB
  float2* pw      = (float2*)(ws + 175865856);   // H*32*32 c64 = 4 MB
  float2* w32     = (float2*)(ws + 180060160);   // H*N c64

  const int TOT = Bsz * Hsz * Lsz;               // 16,777,216

  transpose_in<<<TOT / 256, 256, 0, stream>>>(x_enc, bufA);
  cvt_bf<<<(NLAY * 2 * Hsz * Hsz) / 256, 256, 0, stream>>>(Wo, wo_bf, NLAY * 2 * Hsz * Hsz);
  cvt_bf<<<(Psz * Lsz) / 256, 256, 0, stream>>>(W_out, wout_bf, Psz * Lsz);

  float* cur = bufA;
  float* nxt = bufB;
  for (int layer = 0; layer < NLAY; ++layer) {
    const float* ld = log_dt + layer * Hsz;
    const float* ar = A_re + (size_t)layer * Hsz * Nst;
    const float* ai = A_im + (size_t)layer * Hsz * Nst;
    const float* cr = C_re + (size_t)layer * Hsz * Nst;
    const float* ci = C_im + (size_t)layer * Hsz * Nst;

    s4_tables<<<(Hsz * Nst) / 256, 256, 0, stream>>>(ld, ar, ai, cr, ci, mcar, pw, w32);
    s4_ktab  <<<(Hsz * 32)  / 256, 256, 0, stream>>>(ld, ar, ai, cr, ci, ktab);
    s4_scan  <<<(Bsz * Hsz) / 8,   256, 0, stream>>>(cur, ktab, mcar, pw, w32,
                                                     Dskip + layer * Hsz, ybf);
    glu_gemm <<<dim3(Lsz / 64, Hsz / 64, Bsz), 256, 0, stream>>>(
        wo_bf + (size_t)layer * 2 * Hsz * Hsz, ybf, bo + layer * 2 * Hsz, nxt);

    float* t = cur; cur = nxt; nxt = t;
  }

  cvt_bf<<<TOT / 256, 256, 0, stream>>>(cur, ybf, TOT);
  out_gemm<<<dim3(Hsz / 64, (Psz + 63) / 64, Bsz), 256, 0, stream>>>(
      wout_bf, ybf, b_out, Out);
}
